// NodeGLAM_37288906064221
// MI455X (gfx1250) — matrix operand order repaired and checked by reading, not compiled, not run on hardware
//
#include <hip/hip_runtime.h>

typedef __attribute__((ext_vector_type(16))) _Float16 v16h;
typedef __attribute__((ext_vector_type(8)))  float    v8f;

#define WMMA_F16(a, b, c) \
  __builtin_amdgcn_wmma_f32_16x16x32_f16(false, (a), false, (b), (short)0, (c), false, false)

__device__ __forceinline__ float gelu_exact(float x) {
  return 0.5f * x * (1.0f + erff(x * 0.70710678118654752f));
}

// ---------------------------------------------------------------------------
// Fragment loaders (wave32, 16x16x32 f16 WMMA)
// A (16x32 f16): lane l -> row = l%16, khalf = l/16; elements e<8 -> k = e+8*kh,
//                e>=8 -> k = e+8+8*kh  => two contiguous 8-half (16B) runs.
// B (32x16 f16): lane l -> col = l%16, khalf = l/16; same k order as A.
//                Weights are pre-packed so a fragment is one contiguous 32B read.
// C (16x16 f32): vgpr g -> row = g + 8*(l/16), col = l%16.
// ---------------------------------------------------------------------------
__device__ __forceinline__ v16h load_a_f16(const _Float16* A, int row, int lda,
                                           int k0, int kh) {
  const _Float16* p = A + (size_t)row * lda + k0 + 8 * kh;
  v16h r;
  uint4* pr = reinterpret_cast<uint4*>(&r);
  pr[0] = *reinterpret_cast<const uint4*>(p);
  pr[1] = *reinterpret_cast<const uint4*>(p + 16);
  return r;
}

__device__ __forceinline__ v16h load_a_f32(const float* A, int row, int lda,
                                           int k0, int kh) {
  const float* p = A + (size_t)row * lda + k0 + 8 * kh;
  v16h r;
#pragma unroll
  for (int e = 0; e < 8; e++) r[e] = (_Float16)p[e];
#pragma unroll
  for (int e = 0; e < 8; e++) r[8 + e] = (_Float16)p[16 + e];
  return r;
}

__device__ __forceinline__ v16h load_b_frag(const _Float16* P, int frag, int lane) {
  const _Float16* p = P + ((size_t)frag * 32 + lane) * 16;
  v16h r;
  uint4* pr = reinterpret_cast<uint4*>(&r);
  pr[0] = *reinterpret_cast<const uint4*>(p);
  pr[1] = *reinterpret_cast<const uint4*>(p + 8);
  return r;
}

// ---------------------------------------------------------------------------
// Generic tall-skinny GEMM: Y = gelu(A @ B + bias).
// One wave = MT x 16 rows; block = 8 waves. B fragments preloaded in chunks of
// 4 into distinct registers so loads clause together and WMMAs overlap them.
// ---------------------------------------------------------------------------
template <int KIN, int NOUT, int MT, bool AF32, bool OUTF16>
__global__ __launch_bounds__(256) void gemm_act(const void* __restrict__ Av, int lda,
                                                const _Float16* __restrict__ Bp,
                                                const float* __restrict__ bias,
                                                void* __restrict__ Yv, int ldy,
                                                int nrows) {
  constexpr int KT = KIN / 32, NT = NOUT / 16;
  constexpr int BCH = (NT < 4) ? NT : 4;
  const int lane = threadIdx.x & 31;
  const int wave = threadIdx.x >> 5;
  const int rowBase = (blockIdx.x * 8 + wave) * 16 * MT;
  if (rowBase >= nrows) return;  // wave-uniform: EXEC stays all-ones for WMMA
  const int ln = lane & 15, kh = lane >> 4;
  int arow[MT];
#pragma unroll
  for (int mt = 0; mt < MT; mt++) arow[mt] = min(rowBase + mt * 16 + ln, nrows - 1);

  v8f c[MT][NT];
#pragma unroll
  for (int mt = 0; mt < MT; mt++)
#pragma unroll
    for (int nt = 0; nt < NT; nt++)
#pragma unroll
      for (int g = 0; g < 8; g++) c[mt][nt][g] = 0.0f;

#pragma unroll
  for (int kt = 0; kt < KT; kt++) {
    v16h a[MT];
#pragma unroll
    for (int mt = 0; mt < MT; mt++) {
      if constexpr (AF32)
        a[mt] = load_a_f32((const float*)Av, arow[mt], lda, kt * 32, kh);
      else
        a[mt] = load_a_f16((const _Float16*)Av, arow[mt], lda, kt * 32, kh);
    }
#pragma unroll
    for (int nt0 = 0; nt0 < NT; nt0 += BCH) {
      v16h bf[BCH];
#pragma unroll
      for (int j = 0; j < BCH; j++) bf[j] = load_b_frag(Bp, kt * NT + nt0 + j, lane);
#pragma unroll
      for (int j = 0; j < BCH; j++)
#pragma unroll
        for (int mt = 0; mt < MT; mt++)
          c[mt][nt0 + j] = WMMA_F16(a[mt], bf[j], c[mt][nt0 + j]);
    }
  }

#pragma unroll
  for (int mt = 0; mt < MT; mt++)
#pragma unroll
    for (int nt = 0; nt < NT; nt++) {
      const int n = nt * 16 + ln;
      const float bv = bias[n];
#pragma unroll
      for (int g = 0; g < 8; g++) {
        const int m = rowBase + mt * 16 + g + 8 * kh;
        if (m < nrows) {
          const float v = gelu_exact(c[mt][nt][g] + bv);
          if constexpr (OUTF16)
            ((_Float16*)Yv)[(size_t)m * ldy + n] = (_Float16)v;
          else
            ((float*)Yv)[(size_t)m * ldy + n] = v;
        }
      }
    }
}

// Y (+)= A_f32 @ B  (no activation) — the 7 TAGConv hop GEMMs. MT=2, NT=4.
template <int KIN, int NOUT, int MT>
__global__ __launch_bounds__(256) void gemm_acc(const float* __restrict__ A, int lda,
                                                const _Float16* __restrict__ Bp,
                                                float* __restrict__ Y, int ldy,
                                                int nrows, int accumulate) {
  constexpr int KT = KIN / 32, NT = NOUT / 16;
  constexpr int BCH = (NT < 4) ? NT : 4;
  const int lane = threadIdx.x & 31;
  const int wave = threadIdx.x >> 5;
  const int rowBase = (blockIdx.x * 8 + wave) * 16 * MT;
  if (rowBase >= nrows) return;
  const int ln = lane & 15, kh = lane >> 4;
  int arow[MT];
#pragma unroll
  for (int mt = 0; mt < MT; mt++) arow[mt] = min(rowBase + mt * 16 + ln, nrows - 1);

  v8f c[MT][NT];
  if (accumulate) {
#pragma unroll
    for (int mt = 0; mt < MT; mt++)
#pragma unroll
      for (int nt = 0; nt < NT; nt++)
#pragma unroll
        for (int g = 0; g < 8; g++) {
          const int m = min(rowBase + mt * 16 + g + 8 * kh, nrows - 1);
          c[mt][nt][g] = Y[(size_t)m * ldy + nt * 16 + ln];
        }
  } else {
#pragma unroll
    for (int mt = 0; mt < MT; mt++)
#pragma unroll
      for (int nt = 0; nt < NT; nt++)
#pragma unroll
        for (int g = 0; g < 8; g++) c[mt][nt][g] = 0.0f;
  }

#pragma unroll
  for (int kt = 0; kt < KT; kt++) {
    v16h a[MT];
#pragma unroll
    for (int mt = 0; mt < MT; mt++) a[mt] = load_a_f32(A, arow[mt], lda, kt * 32, kh);
#pragma unroll
    for (int nt0 = 0; nt0 < NT; nt0 += BCH) {
      v16h bf[BCH];
#pragma unroll
      for (int j = 0; j < BCH; j++) bf[j] = load_b_frag(Bp, kt * NT + nt0 + j, lane);
#pragma unroll
      for (int j = 0; j < BCH; j++)
#pragma unroll
        for (int mt = 0; mt < MT; mt++)
          c[mt][nt0 + j] = WMMA_F16(a[mt], bf[j], c[mt][nt0 + j]);
    }
  }

#pragma unroll
  for (int mt = 0; mt < MT; mt++)
#pragma unroll
    for (int nt = 0; nt < NT; nt++) {
      const int n = nt * 16 + ln;
#pragma unroll
      for (int g = 0; g < 8; g++) {
        const int m = rowBase + mt * 16 + g + 8 * kh;
        if (m < nrows) Y[(size_t)m * ldy + n] = c[mt][nt][g];
      }
    }
}

// Final layer: logits = A @ end_w + b, then row softmax (NC=16) via half-wave
// xor-shuffles (masks 1,2,4,8 keep lanes 0-15 / 16-31 independent groups).
template <int KIN>
__global__ __launch_bounds__(256) void gemm_softmax(const _Float16* __restrict__ A,
                                                    int lda,
                                                    const _Float16* __restrict__ Bp,
                                                    const float* __restrict__ bias,
                                                    float* __restrict__ Y, int nrows) {
  constexpr int KT = KIN / 32;
  const int lane = threadIdx.x & 31;
  const int wave = threadIdx.x >> 5;
  const int rowBase = (blockIdx.x * 8 + wave) * 16;
  if (rowBase >= nrows) return;
  const int ln = lane & 15, kh = lane >> 4;
  const int arow = min(rowBase + ln, nrows - 1);

  v8f c;
#pragma unroll
  for (int g = 0; g < 8; g++) c[g] = 0.0f;

  v16h bf[KT];
#pragma unroll
  for (int kt = 0; kt < KT; kt++) bf[kt] = load_b_frag(Bp, kt, lane);
#pragma unroll
  for (int kt = 0; kt < KT; kt++) {
    v16h a = load_a_f16(A, arow, lda, kt * 32, kh);
    c = WMMA_F16(a, bf[kt], c);
  }

  const float bv = bias[ln];
#pragma unroll
  for (int g = 0; g < 8; g++) {
    float x = c[g] + bv;
    float mx = x;
#pragma unroll
    for (int s = 1; s < 16; s <<= 1) mx = fmaxf(mx, __shfl_xor(mx, s, 32));
    const float e = __expf(x - mx);
    float sum = e;
#pragma unroll
    for (int s = 1; s < 16; s <<= 1) sum += __shfl_xor(sum, s, 32);
    const int m = rowBase + g + 8 * kh;
    if (m < nrows) Y[(size_t)m * 16 + ln] = e / sum;
  }
}

// ---------------------------------------------------------------------------
// Weight packing: f32 row-major W[KIN][NOUT] -> f16 B-fragment layout.
// ---------------------------------------------------------------------------
__global__ void pack_w(const float* __restrict__ W, _Float16* __restrict__ dst,
                       int KIN, int NOUT) {
  const int NT = NOUT / 16, KT = KIN / 32;
  const int total = KT * NT * 32;
  const int gid = blockIdx.x * 256 + threadIdx.x;
  if (gid >= total) return;
  const int frag = gid >> 5, lane = gid & 31;
  const int kt = frag / NT, nt = frag % NT;
  const int n = nt * 16 + (lane & 15), kh = lane >> 4;
  _Float16* o = dst + (size_t)gid * 16;
#pragma unroll
  for (int e = 0; e < 16; e++) {
    const int k = kt * 32 + (e < 8 ? e + 8 * kh : e + 8 + 8 * kh);
    o[e] = (_Float16)W[(size_t)k * NOUT + n];
  }
}

// ---------------------------------------------------------------------------
// BatchNorm stats: one block per feature (F=64), fold into scale/shift.
// ---------------------------------------------------------------------------
__global__ void bn_stats(const float* __restrict__ x, const float* __restrict__ gamma,
                         const float* __restrict__ beta, float* __restrict__ scale,
                         float* __restrict__ shift, int n) {
  const int f = blockIdx.x, t = threadIdx.x;
  float s = 0.0f, ss = 0.0f;
  for (int r = t; r < n; r += 256) {
    const float v = x[(size_t)r * 64 + f];
    s += v;
    ss += v * v;
  }
  __shared__ float bs[256], bss[256];
  bs[t] = s;
  bss[t] = ss;
  __syncthreads();
  for (int o = 128; o > 0; o >>= 1) {
    if (t < o) {
      bs[t] += bs[t + o];
      bss[t] += bss[t + o];
    }
    __syncthreads();
  }
  if (t == 0) {
    const float mean = bs[0] / n;
    const float var = bss[0] / n - mean * mean;
    const float sc = gamma[f] * rsqrtf(var + 1e-5f);
    scale[f] = sc;
    shift[f] = beta[f] - mean * sc;
  }
}

// xn (f16) into ain[:, 0:64]  (ain row stride = 128 halves)
__global__ void xn_kernel(const float* __restrict__ x, const float* __restrict__ scale,
                          const float* __restrict__ shift, _Float16* __restrict__ ain,
                          int n) {
  const int idx = blockIdx.x * 256 + threadIdx.x;
  const int r = idx >> 6, f = idx & 63;
  if (r < n) ain[(size_t)r * 128 + f] = (_Float16)(x[(size_t)r * 64 + f] * scale[f] + shift[f]);
}

// h = gelu(out + tag_b) (f16) into ain[:, 64:128]
__global__ void hfinal_kernel(const float* __restrict__ outT,
                              const float* __restrict__ tag_b,
                              _Float16* __restrict__ ain, int n) {
  const int idx = blockIdx.x * 256 + threadIdx.x;
  const int r = idx >> 6, j = idx & 63;
  if (r < n)
    ain[(size_t)r * 128 + 64 + j] = (_Float16)gelu_exact(outT[(size_t)r * 64 + j] + tag_b[j]);
}

__global__ void zero_f32(float* __restrict__ p, int n) {
  const int i = blockIdx.x * 256 + threadIdx.x;
  if (i < n) p[i] = 0.0f;
}

__global__ void degree_kernel(const int* __restrict__ col, int* __restrict__ deg, int E) {
  const int e = blockIdx.x * 256 + threadIdx.x;
  if (e < E) atomicAdd(&deg[col[e]], 1);
}

__global__ void dinv_kernel(const int* __restrict__ deg, float* __restrict__ dinv, int n) {
  const int i = blockIdx.x * 256 + threadIdx.x;
  if (i < n) {
    const int d = deg[i];
    dinv[i] = (d > 0) ? rsqrtf((float)d) : 0.0f;
  }
}

// SpMM hop: one wave per edge; 32 lanes x float4 = full 128-wide row.
// hout[col] += dinv[row]*dinv[col] * hin[row].  h lives in L2 (51MB < 192MB).
__global__ __launch_bounds__(256) void spmm_kernel(const int* __restrict__ row,
                                                   const int* __restrict__ col,
                                                   const float* __restrict__ dinv,
                                                   const float* __restrict__ hin,
                                                   float* __restrict__ hout, int E) {
  const int e = blockIdx.x * 8 + (threadIdx.x >> 5);
  if (e >= E) return;
  const int lane = threadIdx.x & 31;
  const int r = row[e], c = col[e];
  const float nrm = dinv[r] * dinv[c];
  const float4 v = reinterpret_cast<const float4*>(hin + (size_t)r * 128)[lane];
  float* dst = hout + (size_t)c * 128 + lane * 4;
  atomicAdd(dst + 0, nrm * v.x);
  atomicAdd(dst + 1, nrm * v.y);
  atomicAdd(dst + 2, nrm * v.z);
  atomicAdd(dst + 3, nrm * v.w);
}

// ---------------------------------------------------------------------------
extern "C" void kernel_launch(void* const* d_in, const int* in_sizes, int n_in,
                              void* d_out, int out_size, void* d_ws, size_t ws_size,
                              hipStream_t stream) {
  const int N = in_sizes[0] / 64;
  const int E = in_sizes[1] / 2;

  const float* x        = (const float*)d_in[0];
  const int*   ei       = (const int*)d_in[1];
  const float* bn_gamma = (const float*)d_in[2];
  const float* bn_beta  = (const float*)d_in[3];
  const float* taglin_w = (const float*)d_in[4];
  const float* taglin_b = (const float*)d_in[5];
  const float* tag_w    = (const float*)d_in[6];
  const float* tag_b    = (const float*)d_in[7];
  const float* lin0_w   = (const float*)d_in[8];
  const float* lin0_b   = (const float*)d_in[9];
  const float* lin1_w   = (const float*)d_in[10];
  const float* lin1_b   = (const float*)d_in[11];
  const float* cls0_w   = (const float*)d_in[12];
  const float* cls0_b   = (const float*)d_in[13];
  const float* end_w    = (const float*)d_in[14];
  const float* end_b    = (const float*)d_in[15];

  const int* rowp = ei;
  const int* colp = ei + E;
  float* out_f = (float*)d_out;

  // Workspace carve-up (512B aligned).
  char* base = (char*)d_ws;
  size_t off = 0;
  auto carve = [&](size_t bytes) -> char* {
    char* p = base + off;
    off = (off + bytes + 511) & ~(size_t)511;
    return p;
  };
  float*    scale  = (float*)carve(64 * sizeof(float));
  float*    shift  = (float*)carve(64 * sizeof(float));
  float*    dinv   = (float*)carve((size_t)N * sizeof(float));
  int*      deg    = (int*)carve((size_t)N * sizeof(int));
  _Float16* ain    = (_Float16*)carve((size_t)N * 128 * sizeof(_Float16));
  float*    hkA    = (float*)carve((size_t)N * 128 * sizeof(float));
  float*    hkB    = (float*)carve((size_t)N * 128 * sizeof(float));
  float*    outT   = (float*)carve((size_t)N * 64 * sizeof(float));
  _Float16* a1     = (_Float16*)carve((size_t)N * 256 * sizeof(_Float16));
  _Float16* cbuf   = (_Float16*)carve((size_t)N * 128 * sizeof(_Float16));
  _Float16* pw_taglin = (_Float16*)carve(16 * 512 * sizeof(_Float16));     // 64->128
  _Float16* pw_tag    = (_Float16*)carve(7 * 16 * 512 * sizeof(_Float16)); // 7x 128->64
  _Float16* pw_lin0   = (_Float16*)carve(64 * 512 * sizeof(_Float16));     // 128->256
  _Float16* pw_lin1   = (_Float16*)carve(128 * 512 * sizeof(_Float16));    // 256->256
  _Float16* pw_cls0   = (_Float16*)carve(64 * 512 * sizeof(_Float16));     // 256->128
  _Float16* pw_end    = (_Float16*)carve(4 * 512 * sizeof(_Float16));      // 128->16
  (void)ws_size; (void)n_in; (void)out_size;

  auto cdiv = [](int a, int b) { return (a + b - 1) / b; };
  const int gb1 = cdiv(N, 128);  // MT=1 kernels: 128 rows/block
  const int gb2 = cdiv(N, 256);  // MT=2 kernels: 256 rows/block

  // BN stats + apply (fused into f16 store of ain[:, :64])
  bn_stats<<<64, 256, 0, stream>>>(x, bn_gamma, bn_beta, scale, shift, N);
  xn_kernel<<<cdiv(N * 64, 256), 256, 0, stream>>>(x, scale, shift, ain, N);

  // Pack all weight matrices into WMMA B-fragment layout (tiny, once per call).
  pack_w<<<cdiv(16 * 32, 256), 256, 0, stream>>>(taglin_w, pw_taglin, 64, 128);
  for (int k = 0; k < 7; k++)
    pack_w<<<cdiv(16 * 32, 256), 256, 0, stream>>>(tag_w + (size_t)k * 128 * 64,
                                                   pw_tag + (size_t)k * 16 * 512, 128, 64);
  pack_w<<<cdiv(64 * 32, 256), 256, 0, stream>>>(lin0_w, pw_lin0, 128, 256);
  pack_w<<<cdiv(128 * 32, 256), 256, 0, stream>>>(lin1_w, pw_lin1, 256, 256);
  pack_w<<<cdiv(64 * 32, 256), 256, 0, stream>>>(cls0_w, pw_cls0, 256, 128);
  pack_w<<<cdiv(4 * 32, 256), 256, 0, stream>>>(end_w, pw_end, 128, 16);

  // Degrees -> dinv
  zero_f32<<<cdiv(N, 256), 256, 0, stream>>>((float*)deg, N);
  degree_kernel<<<cdiv(E, 256), 256, 0, stream>>>(colp, deg, E);
  dinv_kernel<<<cdiv(N, 256), 256, 0, stream>>>(deg, dinv, N);

  // h0 = gelu(xn @ tag_lin_w + b)  [N,128] f32   (MT=2)
  gemm_act<64, 128, 2, false, false><<<gb2, 256, 0, stream>>>(
      ain, 128, pw_taglin, taglin_b, hkA, 128, N);

  // out = h0 @ tag_w[0]   (MT=2)
  gemm_acc<128, 64, 2><<<gb2, 256, 0, stream>>>(hkA, 128, pw_tag, outT, 64, N, 0);

  // 6 propagation hops: hk = D^-1/2 A D^-1/2 hk;  out += hk @ tag_w[k]
  float* cur = hkA;
  float* nxt = hkB;
  for (int k = 1; k <= 6; k++) {
    zero_f32<<<cdiv(N * 128, 256), 256, 0, stream>>>(nxt, N * 128);
    spmm_kernel<<<cdiv(E, 8), 256, 0, stream>>>(rowp, colp, dinv, cur, nxt, E);
    gemm_acc<128, 64, 2><<<gb2, 256, 0, stream>>>(
        nxt, 128, pw_tag + (size_t)k * 16 * 512, outT, 64, N, 1);
    float* t = cur; cur = nxt; nxt = t;
  }

  // ain[:, 64:128] = gelu(out + tag_b)
  hfinal_kernel<<<cdiv(N * 64, 256), 256, 0, stream>>>(outT, tag_b, ain, N);

  // a = gelu([xn, h] @ lin0 + b) -> gelu(@ lin1 + b)  (second written f32 to d_out)
  gemm_act<128, 256, 1, false, true><<<gb1, 256, 0, stream>>>(
      ain, 128, pw_lin0, lin0_b, a1, 256, N);
  gemm_act<256, 256, 1, false, false><<<gb1, 256, 0, stream>>>(
      a1, 256, pw_lin1, lin1_b, out_f, 256, N);

  // classifier: gelu(a @ cls0 + b) -> softmax(@ end + b)
  gemm_act<256, 128, 2, true, true><<<gb2, 256, 0, stream>>>(
      out_f, 256, pw_cls0, cls0_b, cbuf, 128, N);
  gemm_softmax<128><<<gb1, 256, 0, stream>>>(
      cbuf, 128, pw_end, end_b, out_f + (size_t)N * 256, N);
}
